// AttnScatterDataReadout_56195352101150
// MI455X (gfx1250) — compile-verified, matrix operand order checked by reading, not run on hardware
//
#include <hip/hip_runtime.h>
#include <hip/hip_bf16.h>

// ---------------------------------------------------------------------------
// Types for CDNA5 WMMA (gfx1250, wave32)
// ---------------------------------------------------------------------------
typedef __attribute__((ext_vector_type(16))) __bf16 v16bf;
typedef __attribute__((ext_vector_type(8)))  float  v8f;
typedef __attribute__((ext_vector_type(4)))  unsigned int v4u;
typedef __attribute__((ext_vector_type(8)))  int    v8i;
typedef __attribute__((ext_vector_type(4)))  int    v4i;

union Frag {            // 16x32 bf16 A-frag / 32x16 bf16 B-frag (8 dwords/lane)
    uint4          u[2];
    unsigned short s[16];
    v16bf          v;
};
union Acc {             // 16x16 f32 C/D tile (8 f32/lane)
    v8f   v;
    float f[8];
};
union V8IU { int e[8]; v8i v; };
union V4UU { unsigned int e[4]; v4u v; };

__device__ __forceinline__ v8f wmma_bf16(v16bf a, v16bf b, v8f c) {
    return __builtin_amdgcn_wmma_f32_16x16x32_bf16(
        /*neg_a=*/false, a, /*neg_b=*/false, b,
        /*c_mod=*/(short)0, c, /*reuse_a=*/false, /*reuse_b=*/false);
}

__device__ __forceinline__ unsigned short f2bf(float x) {
    unsigned int u = __float_as_uint(x);
    unsigned int r = u + 0x7FFFu + ((u >> 16) & 1u);   // round-to-nearest-even
    return (unsigned short)(r >> 16);
}

// Load a 16x32 bf16 fragment; rows (row0 + lane&15) of a K-major matrix.
// Lane half h=0 holds K 0..7 & 16..23 (dwords {0..3, 8..11}), h=1 holds
// K 8..15 & 24..31 (dwords {4..7, 12..15}) -> uint4 idx {h, 2+h}.
__device__ __forceinline__ Frag load_frag(const unsigned short* base,
                                          int row0, int k0, int ld, int lane) {
    int r  = lane & 15;
    int hh = lane >> 4;
    const uint4* p = (const uint4*)(base + (size_t)(row0 + r) * ld + k0);
    Frag f;
    f.u[0] = p[hh];
    f.u[1] = p[2 + hh];
    return f;
}

// ---------------------------------------------------------------------------
// TDM (Tensor Data Mover) 2D tile load: global -> LDS.  D# layout per
// CDNA5 ISA sec.8.  Issued once per wave (SGPR operands).
// ---------------------------------------------------------------------------
#if defined(__has_builtin)
#  if __has_builtin(__builtin_amdgcn_tensor_load_to_lds) && \
      __has_builtin(__builtin_amdgcn_s_wait_tensorcnt)
#    define USE_TDM 1
#  endif
#endif
#ifndef USE_TDM
#  define USE_TDM 0
#endif

#if USE_TDM
__device__ __forceinline__ void tdm_load_tile_2d(unsigned lds_off,
                                                 unsigned long long gaddr,
                                                 unsigned tensor_d0, unsigned tensor_d1,
                                                 unsigned tile_d0, unsigned tile_d1,
                                                 unsigned long long stride0_elems) {
    V4UU g0;
    g0.e[0] = 1u;                                        // count=1, user D#
    g0.e[1] = lds_off;                                   // lds_addr (bytes)
    g0.e[2] = (unsigned)(gaddr & 0xFFFFFFFFu);           // global_addr[31:0]
    g0.e[3] = (unsigned)((gaddr >> 32) & 0x1FFFFFFu) | (2u << 30); // addr hi | type=2
    V8IU g1;
    g1.e[0] = (int)(1u << 16);                           // data_size=1 (2 bytes)
    g1.e[1] = (int)(tensor_d0 << 16);                    // tensor_dim0[15:0] @bits63:48
    g1.e[2] = (int)((tensor_d0 >> 16) | (tensor_d1 << 16)); // dim0 hi | dim1 lo
    g1.e[3] = (int)(((tensor_d1 >> 16) & 0xFFFFu) | (tile_d0 << 16)); // dim1 hi|tile0
    g1.e[4] = (int)(tile_d1 & 0xFFFFu);                  // tile_dim1 (tile_dim2=0)
    g1.e[5] = (int)(unsigned)(stride0_elems & 0xFFFFFFFFu);     // dim0 stride lo
    g1.e[6] = (int)(unsigned)((stride0_elems >> 32) & 0xFFFFu); // dim0 stride hi
    g1.e[7] = 0;
    v4i z4 = {0, 0, 0, 0};
#if defined(__clang_major__) && (__clang_major__ >= 23)
    v8i z8 = {0, 0, 0, 0, 0, 0, 0, 0};
    __builtin_amdgcn_tensor_load_to_lds(g0.v, g1.v, z4, z4, z8, 0);
#else
    __builtin_amdgcn_tensor_load_to_lds(g0.v, g1.v, z4, z4, 0);
#endif
}
#endif

// ---------------------------------------------------------------------------
// Problem constants
// ---------------------------------------------------------------------------
#define BC_   32            // B*C
#define NTOK_ 1024          // N
#define DMODEL_ 256         // D
#define NH_   8             // H
#define DH_   32            // D/H
#define HID_  768
#define LLOG_ 256           // L
#define T1_   ((size_t)BC_ * NTOK_)   // 32768 tokens
#define T2_   ((size_t)BC_ * LLOG_)   // 8192 tokens

// ---------------------------------------------------------------------------
// Elementwise / reduction kernels
// ---------------------------------------------------------------------------
__device__ __forceinline__ float block_reduce_sum_256(float v) {
    __shared__ float s[8];
#pragma unroll
    for (int m = 16; m >= 1; m >>= 1) v += __shfl_xor(v, m, 32);
    int wid = threadIdx.x >> 5;
    if ((threadIdx.x & 31) == 0) s[wid] = v;
    __syncthreads();
    float tot = 0.f;
#pragma unroll
    for (int i = 0; i < 8; ++i) tot += s[i];
    return tot;
}

__global__ void k_cvt_bf16(const float* __restrict__ in,
                           unsigned short* __restrict__ out, int n) {
    int i = blockIdx.x * 256 + threadIdx.x;
    if (i < n) out[i] = f2bf(in[i]);
}

__global__ void k_rms(const float* __restrict__ x, const float* __restrict__ w,
                      unsigned short* __restrict__ out) {
    size_t t = blockIdx.x;
    float v  = x[t * DMODEL_ + threadIdx.x];
    float ss = block_reduce_sum_256(v * v);
    float sc = rsqrtf(ss * (1.0f / DMODEL_) + 1e-5f);
    out[t * DMODEL_ + threadIdx.x] = f2bf(v * sc * w[threadIdx.x]);
}

// h13 is [tokens][1536] = [w1 row | w3 row]; g = silu(h1)*h3 -> bf16 [tokens][768]
__global__ void k_silumul2(const float* __restrict__ h13,
                           unsigned short* __restrict__ g, size_t n) {
    size_t i = (size_t)blockIdx.x * 256 + threadIdx.x;
    if (i < n) {
        size_t t = i / HID_;
        int    j = (int)(i - t * HID_);
        float a = h13[t * (2 * HID_) + j];
        float b = h13[t * (2 * HID_) + HID_ + j];
        float s = a / (1.0f + __expf(-a));
        g[i] = f2bf(s * b);
    }
}

__global__ void k_scatter_mul(const float* __restrict__ x, const int* __restrict__ map,
                              float* __restrict__ lb, int E) {
    int bc = blockIdx.x;
    int d  = threadIdx.x;
    float*       col = lb + (size_t)bc * LLOG_ * DMODEL_ + d;
    const float* xb  = x + (size_t)bc * NTOK_ * DMODEL_ + d;
    for (int j = 0; j < LLOG_; ++j) col[(size_t)j * DMODEL_] = 1.0f;
    for (int e = 0; e < E; ++e) {
        int s = map[e];
        int t = map[E + e];
        col[(size_t)t * DMODEL_] *= xb[(size_t)s * DMODEL_];
    }
}

__global__ void k_head(const float* __restrict__ l, const float* __restrict__ w,
                       const float* __restrict__ b, float* __restrict__ out) {
    size_t t = blockIdx.x;
    float v  = l[t * DMODEL_ + threadIdx.x] * w[threadIdx.x];
    float s  = block_reduce_sum_256(v);
    if (threadIdx.x == 0) out[t] = s + b[0];
}

// ---------------------------------------------------------------------------
// WMMA GEMM with double-buffered TDM-staged LDS tiles:
//   out[M,N] = A[M,K](bf16) * W[N,K](bf16)^T (+bias)(+resid)(act)
// block = 256 thr (8 waves, 4x2), block tile 128x128, K-step 32.
// Wave 0 issues two TDM tile DMAs (A:128x32, B:128x32) for stage s+1 into the
// alternate LDS buffer, then s_wait_tensorcnt(2) so only stage s must have
// landed (TDM ops complete in order per wave) -> DMA overlaps WMMA compute.
// ---------------------------------------------------------------------------
__global__ __launch_bounds__(256) void k_gemm_bf16(
    const unsigned short* __restrict__ A, const unsigned short* __restrict__ W,
    const float* __restrict__ bias, const float* __restrict__ resid,
    float* __restrict__ outF, unsigned short* __restrict__ outBF,
    int M, int N, int K, int act) {
    (void)M;
    __shared__ __align__(16) unsigned short aT[2][128 * 32];   // 2 x 8 KB
    __shared__ __align__(16) unsigned short bT[2][128 * 32];   // 2 x 8 KB
    int lane = threadIdx.x & 31;
    int wid  = threadIdx.x >> 5;
    int mBlk = blockIdx.x * 128;
    int nBlk = blockIdx.y * 128;
    int m0 = (wid & 3) * 32;     // local row tile in aT
    int n0 = (wid >> 2) * 64;    // local col tile in bT

    Acc acc[2][4];
#pragma unroll
    for (int mi = 0; mi < 2; ++mi)
#pragma unroll
        for (int ni = 0; ni < 4; ++ni)
#pragma unroll
            for (int r = 0; r < 8; ++r) acc[mi][ni].f[r] = 0.f;

    const int nst = K / 32;

#if USE_TDM
    auto issue = [&](int s, int buf) {
        int k0 = s * 32;
        tdm_load_tile_2d((unsigned)(size_t)&aT[buf][0],
                         (unsigned long long)(size_t)(A + (size_t)mBlk * K + k0),
                         (unsigned)K, 1u << 20, 32u, 128u, (unsigned long long)K);
        tdm_load_tile_2d((unsigned)(size_t)&bT[buf][0],
                         (unsigned long long)(size_t)(W + (size_t)nBlk * K + k0),
                         (unsigned)K, 1u << 20, 32u, 128u, (unsigned long long)K);
    };
    if (wid == 0) issue(0, 0);
#endif

    for (int s = 0; s < nst; ++s) {
        const int cur = s & 1;
#if USE_TDM
        if (wid == 0) {
            if (s + 1 < nst) {
                issue(s + 1, cur ^ 1);                     // prefetch next stage
                __builtin_amdgcn_s_wait_tensorcnt((short)2); // stage s landed
            } else {
                __builtin_amdgcn_s_wait_tensorcnt((short)0);
            }
        }
        __syncthreads();
#else
        {
            int k0 = s * 32;
            int t  = threadIdx.x;
#pragma unroll
            for (int i = t; i < 512; i += 256) {            // 512 x uint4 each tile
                int r = i >> 2, q = i & 3;
                ((uint4*)aT[cur])[i] = *((const uint4*)(A + (size_t)(mBlk + r) * K + k0) + q);
                ((uint4*)bT[cur])[i] = *((const uint4*)(W + (size_t)(nBlk + r) * K + k0) + q);
            }
        }
        __syncthreads();
#endif
        const unsigned short* aP = aT[cur];
        const unsigned short* bP = bT[cur];
        Frag a0 = load_frag(aP, m0, 0, 32, lane);
        Frag a1 = load_frag(aP, m0 + 16, 0, 32, lane);
#pragma unroll
        for (int ni = 0; ni < 4; ++ni) {
            Frag b = load_frag(bP, n0 + ni * 16, 0, 32, lane);
            acc[0][ni].v = wmma_bf16(a0.v, b.v, acc[0][ni].v);
            acc[1][ni].v = wmma_bf16(a1.v, b.v, acc[1][ni].v);
        }
        __syncthreads();
    }

    int cl = lane & 15, hh = lane >> 4;
#pragma unroll
    for (int mi = 0; mi < 2; ++mi) {
#pragma unroll
        for (int ni = 0; ni < 4; ++ni) {
            int col  = nBlk + n0 + ni * 16 + cl;
            int rowb = mBlk + m0 + mi * 16 + hh * 8;
            float bv = bias ? bias[col] : 0.f;
#pragma unroll
            for (int r = 0; r < 8; ++r) {
                size_t o = (size_t)(rowb + r) * N + col;
                float v  = acc[mi][ni].f[r] + bv;
                if (resid) v += resid[o];
                if (act == 1) v = tanhf(v);
                if (outF) outF[o] = v;
                else      outBF[o] = f2bf(v);
            }
        }
    }
}

// ---------------------------------------------------------------------------
// Flash attention, dh=32: 1 wave per (bc*H, 16-row q tile).
// qkv layout: [token][3*D], q at +0, k at +D, v at +2D, head h at +h*32.
// ---------------------------------------------------------------------------
__global__ __launch_bounds__(32) void k_flash(const unsigned short* __restrict__ qkv,
                                              unsigned short* __restrict__ obf) {
    const int lane = threadIdx.x;
    const int cl = lane & 15, hh = lane >> 4;
    const int bh = blockIdx.x;
    const int bc = bh / NH_, h = bh % NH_;
    const int qt = blockIdx.y;
    const int TD = 3 * DMODEL_;                 // 768
    const size_t baseTok = (size_t)bc * NTOK_;

    Frag qf;
    {
        const uint4* p = (const uint4*)(qkv + (baseTok + qt * 16 + cl) * TD + h * DH_);
        qf.u[0] = p[hh];
        qf.u[1] = p[2 + hh];
    }

    float m[8], lsum[8], c[8];
    Acc o0, o1;
#pragma unroll
    for (int r = 0; r < 8; ++r) { m[r] = -1e30f; lsum[r] = 0.f; o0.f[r] = 0.f; o1.f[r] = 0.f; }
    const float scale = 0.17677669529663687f;   // 1/sqrt(32)

    __shared__ __align__(16) unsigned short pl[16 * 32];

    for (int kv0 = 0; kv0 < NTOK_; kv0 += 32) {
        Frag kf0, kf1;
        {
            const uint4* p = (const uint4*)(qkv + (baseTok + kv0 + cl) * TD + DMODEL_ + h * DH_);
            kf0.u[0] = p[hh]; kf0.u[1] = p[2 + hh];
        }
        {
            const uint4* p = (const uint4*)(qkv + (baseTok + kv0 + 16 + cl) * TD + DMODEL_ + h * DH_);
            kf1.u[0] = p[hh]; kf1.u[1] = p[2 + hh];
        }
        v8f z = {};
        Acc s0, s1;
        s0.v = wmma_bf16(qf.v, kf0.v, z);
        s1.v = wmma_bf16(qf.v, kf1.v, z);

#pragma unroll
        for (int r = 0; r < 8; ++r) {
            float a = s0.f[r] * scale, b = s1.f[r] * scale;
            s0.f[r] = a; s1.f[r] = b;
            float t = fmaxf(a, b);
#pragma unroll
            for (int msk = 1; msk <= 8; msk <<= 1) t = fmaxf(t, __shfl_xor(t, msk, 32));
            float mn = fmaxf(m[r], t);
            c[r] = __expf(m[r] - mn);
            m[r] = mn;
        }
#pragma unroll
        for (int r = 0; r < 8; ++r) {
            float p0 = __expf(s0.f[r] - m[r]);
            float p1 = __expf(s1.f[r] - m[r]);
            s0.f[r] = p0; s1.f[r] = p1;
            float t = p0 + p1;
#pragma unroll
            for (int msk = 1; msk <= 8; msk <<= 1) t += __shfl_xor(t, msk, 32);
            lsum[r] = lsum[r] * c[r] + t;
            o0.f[r] *= c[r];
            o1.f[r] *= c[r];
        }

        // C-layout -> A-layout transpose of P through LDS
#pragma unroll
        for (int r = 0; r < 8; ++r) {
            int row = hh * 8 + r;
            pl[row * 32 + cl]      = f2bf(s0.f[r]);
            pl[row * 32 + 16 + cl] = f2bf(s1.f[r]);
        }
        __syncthreads();
        Frag pf;
        {
            const uint4* p = (const uint4*)(pl + cl * 32);
            pf.u[0] = p[hh];
            pf.u[1] = p[2 + hh];
        }
        __syncthreads();

        // V fragments (B matrix 32x16: column = dh feature -> strided gather)
        Frag vf0, vf1;
        {
            const unsigned short* vb = qkv + (baseTok + kv0) * TD + 2 * DMODEL_ + h * DH_;
#pragma unroll
            for (int vv = 0; vv < 8; ++vv) {
                int kk = (vv < 4) ? (hh * 8 + 2 * vv) : (16 + hh * 8 + 2 * (vv - 4));
                vf0.s[2 * vv]     = vb[(size_t)kk * TD + cl];
                vf0.s[2 * vv + 1] = vb[(size_t)(kk + 1) * TD + cl];
                vf1.s[2 * vv]     = vb[(size_t)kk * TD + 16 + cl];
                vf1.s[2 * vv + 1] = vb[(size_t)(kk + 1) * TD + 16 + cl];
            }
        }
        o0.v = wmma_bf16(pf.v, vf0.v, o0.v);
        o1.v = wmma_bf16(pf.v, vf1.v, o1.v);
    }

#pragma unroll
    for (int r = 0; r < 8; ++r) {
        float inv = 1.0f / lsum[r];
        size_t rb = (baseTok + qt * 16 + hh * 8 + r) * DMODEL_ + h * DH_;
        obf[rb + cl]      = f2bf(o0.f[r] * inv);
        obf[rb + 16 + cl] = f2bf(o1.f[r] * inv);
    }
}

// ---------------------------------------------------------------------------
// Host orchestration
// ---------------------------------------------------------------------------
extern "C" void kernel_launch(void* const* d_in, const int* in_sizes, int n_in,
                              void* d_out, int out_size, void* d_ws, size_t ws_size,
                              hipStream_t stream) {
    (void)n_in; (void)out_size; (void)ws_size;
    const float* v        = (const float*)d_in[0];
    const int*   map      = (const int*)d_in[1];
    const float* attn_nw  = (const float*)d_in[2];
    const float* in_w     = (const float*)d_in[3];
    const float* in_b     = (const float*)d_in[4];
    const float* out_w    = (const float*)d_in[5];
    const float* out_b    = (const float*)d_in[6];
    const float* ffn_nw   = (const float*)d_in[7];
    const float* ffn_w1   = (const float*)d_in[8];
    const float* ffn_w2   = (const float*)d_in[9];
    const float* ffn_w3   = (const float*)d_in[10];
    const float* du_nw    = (const float*)d_in[11];
    const float* du_w1    = (const float*)d_in[12];
    const float* du_w2    = (const float*)d_in[13];
    const float* du_w3    = (const float*)d_in[14];
    const float* lu_nw    = (const float*)d_in[15];
    const float* lu_w1    = (const float*)d_in[16];
    const float* lu_w2    = (const float*)d_in[17];
    const float* lu_w3    = (const float*)d_in[18];
    const float* head_w   = (const float*)d_in[19];
    const float* head_b   = (const float*)d_in[20];
    float* outp = (float*)d_out;
    const int E = in_sizes[1] / 2;

    char* ws = (char*)d_ws;
    size_t off = 0;
    auto take = [&](size_t bytes) -> char* {
        char* p = ws + off;
        off = (off + bytes + 255) & ~(size_t)255;
        return p;
    };
    const size_t WSZ = (size_t)HID_ * DMODEL_;   // 196608 elements per big weight
    unsigned short* w_qkv  = (unsigned short*)take(WSZ * 2);
    unsigned short* w_out  = (unsigned short*)take((size_t)DMODEL_ * DMODEL_ * 2);
    unsigned short* w_f13  = (unsigned short*)take(2 * WSZ * 2);  // [w1 ; w3]
    unsigned short* w_f2   = (unsigned short*)take(WSZ * 2);
    unsigned short* w_d13  = (unsigned short*)take(2 * WSZ * 2);
    unsigned short* w_d2   = (unsigned short*)take(WSZ * 2);
    unsigned short* w_l13  = (unsigned short*)take(2 * WSZ * 2);
    unsigned short* w_l2   = (unsigned short*)take(WSZ * 2);
    unsigned short* xn     = (unsigned short*)take(T1_ * DMODEL_ * 2);
    unsigned short* qkv    = (unsigned short*)take(T1_ * 3 * DMODEL_ * 2);
    unsigned short* attn   = (unsigned short*)take(T1_ * DMODEL_ * 2);
    float*          rA     = (float*)take(T1_ * DMODEL_ * 4);
    float*          rB     = (float*)take(T1_ * DMODEL_ * 4);
    float*          h13    = (float*)take(T1_ * 2 * HID_ * 4);
    unsigned short* gbf    = (unsigned short*)take(T1_ * HID_ * 2);
    float*          lbuf   = (float*)take(T2_ * DMODEL_ * 4);
    float*          lfin   = (float*)take(T2_ * DMODEL_ * 4);

    auto cvt = [&](const float* src, unsigned short* dst, int n) {
        k_cvt_bf16<<<(n + 255) / 256, 256, 0, stream>>>(src, dst, n);
    };
    auto gemm = [&](const unsigned short* A, const unsigned short* W,
                    const float* bias, const float* resid,
                    float* oF, unsigned short* oB, int M, int N, int K, int act) {
        dim3 g(M / 128, N / 128);
        k_gemm_bf16<<<g, 256, 0, stream>>>(A, W, bias, resid, oF, oB, M, N, K, act);
    };

    // Weight conversion (f32 -> bf16); w1||w3 packed contiguously per stack.
    cvt(in_w, w_qkv, (int)WSZ);
    cvt(out_w, w_out, DMODEL_ * DMODEL_);
    cvt(ffn_w1, w_f13, (int)WSZ); cvt(ffn_w3, w_f13 + WSZ, (int)WSZ); cvt(ffn_w2, w_f2, (int)WSZ);
    cvt(du_w1, w_d13, (int)WSZ);  cvt(du_w3, w_d13 + WSZ, (int)WSZ);  cvt(du_w2, w_d2, (int)WSZ);
    cvt(lu_w1, w_l13, (int)WSZ);  cvt(lu_w3, w_l13 + WSZ, (int)WSZ);  cvt(lu_w2, w_l2, (int)WSZ);

    // --- Attention block: a = mha(rms(x)) + x ---
    k_rms<<<(unsigned)T1_, 256, 0, stream>>>(v, attn_nw, xn);
    gemm(xn, w_qkv, in_b, nullptr, nullptr, qkv, (int)T1_, 3 * DMODEL_, DMODEL_, 0);
    k_flash<<<dim3(BC_ * NH_, NTOK_ / 16), 32, 0, stream>>>(qkv, attn);
    gemm(attn, w_out, out_b, v, rA, nullptr, (int)T1_, DMODEL_, DMODEL_, 0);

    // --- FFN: x = swiglu(rms(a)) + a ---
    k_rms<<<(unsigned)T1_, 256, 0, stream>>>(rA, ffn_nw, xn);
    gemm(xn, w_f13, nullptr, nullptr, h13, nullptr, (int)T1_, 2 * HID_, DMODEL_, 0);
    k_silumul2<<<(unsigned)(T1_ * HID_ / 256), 256, 0, stream>>>(h13, gbf, T1_ * HID_);
    gemm(gbf, w_f2, nullptr, rA, rB, nullptr, (int)T1_, DMODEL_, HID_, 0);

    // --- DU: x = tanh(swiglu(rms(x)) + x) ---
    k_rms<<<(unsigned)T1_, 256, 0, stream>>>(rB, du_nw, xn);
    gemm(xn, w_d13, nullptr, nullptr, h13, nullptr, (int)T1_, 2 * HID_, DMODEL_, 0);
    k_silumul2<<<(unsigned)(T1_ * HID_ / 256), 256, 0, stream>>>(h13, gbf, T1_ * HID_);
    gemm(gbf, w_d2, nullptr, rB, rA, nullptr, (int)T1_, DMODEL_, HID_, 1 /*tanh*/);

    // --- Scatter-multiply readout buffer ---
    k_scatter_mul<<<BC_, 256, 0, stream>>>(rA, map, lbuf, E);

    // --- LU: l = swiglu(rms(l)) + l ---
    k_rms<<<(unsigned)T2_, 256, 0, stream>>>(lbuf, lu_nw, xn);
    gemm(xn, w_l13, nullptr, nullptr, h13, nullptr, (int)T2_, 2 * HID_, DMODEL_, 0);
    k_silumul2<<<(unsigned)(T2_ * HID_ / 256), 256, 0, stream>>>(h13, gbf, T2_ * HID_);
    gemm(gbf, w_l2, nullptr, lbuf, lfin, nullptr, (int)T2_, DMODEL_, HID_, 0);

    // --- Head ---
    k_head<<<(unsigned)T2_, 256, 0, stream>>>(lfin, head_w, head_b, outp);
}
